// MultiheadAttention_24249385353229
// MI455X (gfx1250) — matrix operand order repaired and checked by reading, not compiled, not run on hardware
//
#include <hip/hip_runtime.h>

#define B_   4
#define T_   2048
#define D_   1024
#define NH_  16
#define HS_  64
#define BIAS_W (2*T_ - 1)   /* 4095 */

typedef __attribute__((ext_vector_type(16))) _Float16 v16h;
typedef __attribute__((ext_vector_type(8)))  _Float16 v8h;
typedef __attribute__((ext_vector_type(8)))  float    v8f;

union FragU { v16h v; v8h h[2]; };

// A-matrix 16x32 f16 fragment: lanes 0-15 hold K=[0..7],[16..23]; lanes 16-31 K=[8..15],[24..31]
__device__ __forceinline__ v16h frag_a(const _Float16* __restrict__ row_ptr, int k, int grp) {
  FragU u;
  u.h[0] = *(const v8h*)(row_ptr + k + grp * 8);
  u.h[1] = *(const v8h*)(row_ptr + k + 16 + grp * 8);
  return u.v;
}
// B-matrix 32x16 f16 fragment: lanes 0-15 hold K=[0..7],[16..23]; lanes 16-31 K=[8..15],[24..31] (col = lane&15)
__device__ __forceinline__ v16h frag_b(const _Float16* __restrict__ col_ptr, int k, int grp) {
  FragU u;
  u.h[0] = *(const v8h*)(col_ptr + k + grp * 8);
  u.h[1] = *(const v8h*)(col_ptr + k + 16 + grp * 8);
  return u.v;
}
__device__ __forceinline__ v8f wmma16(v16h a, v16h b, v8f c) {
  return __builtin_amdgcn_wmma_f32_16x16x32_f16(false, a, false, b, (short)0, c, false, false);
}

__global__ void f32_to_f16_kernel(const float* __restrict__ src, _Float16* __restrict__ dst, int n) {
  int i = blockIdx.x * blockDim.x + threadIdx.x;
  if (i < n) dst[i] = (_Float16)src[i];
}

// T5 relative bias table: tab[h][delta + T-1], delta = key - query in [-(T-1), T-1]
__global__ void bias_table_kernel(const float* __restrict__ rel_emb, float* __restrict__ tab) {
  int idx = blockIdx.x * blockDim.x + threadIdx.x;
  int total = NH_ * BIAS_W;
  if (idx >= total) return;
  int h = idx / BIAS_W;
  int d = (idx % BIAS_W) - (T_ - 1);
  int bucket = (d > 0) ? 16 : 0;
  int rp = (d < 0) ? -d : d;
  int val;
  if (rp < 8) {
    val = rp;
  } else {
    float f = __logf((float)rp / 7.0f) / __logf(128.0f / 7.0f) * 8.0f;
    val = 7 + (int)f;
    if (val > 15) val = 15;
  }
  tab[idx] = rel_emb[(bucket + val) * NH_ + h];
}

// C = X(MxK) @ W^T (W is [N,K] row-major).  Block tile 128x128, 8 waves (4M x 2N),
// wave tile 32x64.  Software-pipelined with an epilogue: the loop only ever loads
// fragments at k+32 with plain immediate offsets (no wrap/select), keeping clause
// addressing clean and avoiding register aliasing between addresses and fragments.
// mode 0: z=blockIdx.z selects Wq/Wk/Wv; writes head-split f16 (Q scaled by 1/8, V transposed).
// mode 1: writes f32 to outF.
__global__ __launch_bounds__(256)
void gemm_qkvo_kernel(const _Float16* __restrict__ X,
                      const _Float16* __restrict__ Wbase,
                      _Float16* __restrict__ Qh,
                      _Float16* __restrict__ Kh,
                      _Float16* __restrict__ Vt,
                      float* __restrict__ outF,
                      int mode)
{
  const int lane = threadIdx.x & 31;
  const int wave = threadIdx.x >> 5;
  const int grp  = lane >> 4;
  const int ln   = lane & 15;
  const int wm = wave & 3, wn = wave >> 2;
  const int Mbase = blockIdx.y * 128 + wm * 32;
  const int Nbase = blockIdx.x * 128 + wn * 64;
  const int z = blockIdx.z;
  const _Float16* W = Wbase + (size_t)z * D_ * D_;

  const _Float16* a0p = X + (size_t)(Mbase + ln) * D_;
  const _Float16* a1p = a0p + 16 * D_;
  const _Float16* bp0 = W + (size_t)(Nbase + ln) * D_;
  const _Float16* bp1 = bp0 + 16 * D_;
  const _Float16* bp2 = bp0 + 32 * D_;
  const _Float16* bp3 = bp0 + 48 * D_;

  v8f acc[2][4] = {};

  // pipeline prologue: fragments for k = 0
  v16h a0 = frag_a(a0p, 0, grp);
  v16h a1 = frag_a(a1p, 0, grp);
  v16h b0 = frag_b(bp0, 0, grp);
  v16h b1 = frag_b(bp1, 0, grp);
  v16h b2 = frag_b(bp2, 0, grp);
  v16h b3 = frag_b(bp3, 0, grp);

  for (int k = 0; k < D_ - 32; k += 32) {
    const int kn = k + 32;
    v16h na0 = frag_a(a0p, kn, grp);
    v16h na1 = frag_a(a1p, kn, grp);
    v16h nb0 = frag_b(bp0, kn, grp);
    v16h nb1 = frag_b(bp1, kn, grp);
    v16h nb2 = frag_b(bp2, kn, grp);
    v16h nb3 = frag_b(bp3, kn, grp);

    acc[0][0] = wmma16(a0, b0, acc[0][0]);
    acc[1][0] = wmma16(a1, b0, acc[1][0]);
    acc[0][1] = wmma16(a0, b1, acc[0][1]);
    acc[1][1] = wmma16(a1, b1, acc[1][1]);
    acc[0][2] = wmma16(a0, b2, acc[0][2]);
    acc[1][2] = wmma16(a1, b2, acc[1][2]);
    acc[0][3] = wmma16(a0, b3, acc[0][3]);
    acc[1][3] = wmma16(a1, b3, acc[1][3]);

    a0 = na0; a1 = na1;
    b0 = nb0; b1 = nb1; b2 = nb2; b3 = nb3;

    __builtin_prefetch(a0p + k + 288, 0, 3);
    __builtin_prefetch(bp0 + k + 288, 0, 3);
  }

  // pipeline epilogue: last k-step
  acc[0][0] = wmma16(a0, b0, acc[0][0]);
  acc[1][0] = wmma16(a1, b0, acc[1][0]);
  acc[0][1] = wmma16(a0, b1, acc[0][1]);
  acc[1][1] = wmma16(a1, b1, acc[1][1]);
  acc[0][2] = wmma16(a0, b2, acc[0][2]);
  acc[1][2] = wmma16(a1, b2, acc[1][2]);
  acc[0][3] = wmma16(a0, b3, acc[0][3]);
  acc[1][3] = wmma16(a1, b3, acc[1][3]);

  #pragma unroll
  for (int mt = 0; mt < 2; ++mt)
  #pragma unroll
  for (int nt = 0; nt < 4; ++nt)
  #pragma unroll
  for (int r = 0; r < 8; ++r) {
    int m = Mbase + mt * 16 + r + 8 * grp;   // C layout: row = r + 8*(lane/16)
    int n = Nbase + nt * 16 + ln;            //           col = lane & 15
    float val = acc[mt][nt][r];
    if (mode == 0) {
      int b = m >> 11, t = m & (T_ - 1);
      int h = n >> 6,  hs = n & (HS_ - 1);
      int bh = b * NH_ + h;
      if (z == 0)      Qh[((size_t)(bh * T_ + t)) * HS_ + hs] = (_Float16)(val * 0.125f);
      else if (z == 1) Kh[((size_t)(bh * T_ + t)) * HS_ + hs] = (_Float16)val;
      else             Vt[((size_t)(bh * HS_ + hs)) * T_ + t] = (_Float16)val;
    } else {
      outF[(size_t)m * D_ + n] = val;
    }
  }
}

// Flash attention per (b,h): 4 waves/block, 16 query rows per wave, 64-key tiles.
__global__ __launch_bounds__(128)
void attn_kernel(const _Float16* __restrict__ Qh,
                 const _Float16* __restrict__ Kh,
                 const _Float16* __restrict__ Vt,
                 const float* __restrict__ bias_tab,
                 _Float16* __restrict__ Oh)
{
  const int lane = threadIdx.x & 31;
  const int wave = threadIdx.x >> 5;
  const int grp = lane >> 4, ln = lane & 15;
  const int bh = blockIdx.y;
  const int b = bh >> 4, h = bh & 15;
  const int qbase = blockIdx.x * 64 + wave * 16;

  __shared__ __align__(16) _Float16 Pst[4][16][72];  // wave-private P staging (C->A relayout)

  const _Float16* Qp = Qh + (size_t)bh * T_ * HS_;
  const _Float16* Kp = Kh + (size_t)bh * T_ * HS_;
  const _Float16* Vp = Vt + (size_t)bh * HS_ * T_;
  const float* btab = bias_tab + h * BIAS_W + (T_ - 1);

  const _Float16* qrow = Qp + (size_t)(qbase + ln) * HS_;
  v16h qf0 = frag_a(qrow, 0, grp);
  v16h qf1 = frag_a(qrow, 32, grp);

  float mrow[8], lrow[8];
  v8f oacc[4] = {};
  #pragma unroll
  for (int r = 0; r < 8; ++r) { mrow[r] = -1e30f; lrow[r] = 0.0f; }

  for (int j = 0; j < T_; j += 64) {
    // S = (Q/8) K^T + bias   (16 x 64 per wave)
    v8f S[4];
    #pragma unroll
    for (int nt = 0; nt < 4; ++nt) {
      const _Float16* kcol = Kp + (size_t)(j + nt * 16 + ln) * HS_;
      v16h kf0 = frag_b(kcol, 0, grp);
      v16h kf1 = frag_b(kcol, 32, grp);
      v8f s = {};
      s = wmma16(qf0, kf0, s);
      s = wmma16(qf1, kf1, s);
      #pragma unroll
      for (int r = 0; r < 8; ++r) {
        int m = qbase + r + 8 * grp;
        int n = j + nt * 16 + ln;
        s[r] += btab[n - m];
      }
      S[nt] = s;
    }
    // online softmax (row reductions across 16 lanes of each half-wave)
    #pragma unroll
    for (int r = 0; r < 8; ++r) {
      float v = fmaxf(fmaxf(S[0][r], S[1][r]), fmaxf(S[2][r], S[3][r]));
      v = fmaxf(v, __shfl_xor(v, 1, 16));
      v = fmaxf(v, __shfl_xor(v, 2, 16));
      v = fmaxf(v, __shfl_xor(v, 4, 16));
      v = fmaxf(v, __shfl_xor(v, 8, 16));
      float newm = fmaxf(mrow[r], v);
      float fac = __expf(mrow[r] - newm);
      mrow[r] = newm;
      lrow[r] *= fac;
      #pragma unroll
      for (int nt = 0; nt < 4; ++nt) oacc[nt][r] *= fac;
      float rowsum = 0.0f;
      #pragma unroll
      for (int nt = 0; nt < 4; ++nt) {
        float p = __expf(S[nt][r] - newm);
        S[nt][r] = p;
        rowsum += p;
      }
      rowsum += __shfl_xor(rowsum, 1, 16);
      rowsum += __shfl_xor(rowsum, 2, 16);
      rowsum += __shfl_xor(rowsum, 4, 16);
      rowsum += __shfl_xor(rowsum, 8, 16);
      lrow[r] += rowsum;
      #pragma unroll
      for (int nt = 0; nt < 4; ++nt)
        Pst[wave][r + 8 * grp][nt * 16 + ln] = (_Float16)S[nt][r];
    }
    // O += P @ V  (P as A-frag from LDS; V^T gives contiguous B-frags)
    #pragma unroll
    for (int ks = 0; ks < 2; ++ks) {
      FragU pu;
      pu.h[0] = *(const v8h*)&Pst[wave][ln][ks * 32 + grp * 8];
      pu.h[1] = *(const v8h*)&Pst[wave][ln][ks * 32 + 16 + grp * 8];
      v16h pf = pu.v;
      #pragma unroll
      for (int nt = 0; nt < 4; ++nt) {
        const _Float16* vcol = Vp + (size_t)(nt * 16 + ln) * T_;
        v16h vf = frag_b(vcol, j + ks * 32, grp);
        oacc[nt] = wmma16(pf, vf, oacc[nt]);
      }
    }
  }
  // normalize and store O as f16 [B,T,D]
  #pragma unroll
  for (int nt = 0; nt < 4; ++nt)
  #pragma unroll
  for (int r = 0; r < 8; ++r) {
    int m = qbase + r + 8 * grp;
    float val = oacc[nt][r] / lrow[r];
    Oh[(size_t)(b * T_ + m) * D_ + h * HS_ + nt * 16 + ln] = (_Float16)val;
  }
}

extern "C" void kernel_launch(void* const* d_in, const int* in_sizes, int n_in,
                              void* d_out, int out_size, void* d_ws, size_t ws_size,
                              hipStream_t stream)
{
  const float* q   = (const float*)d_in[0];
  const float* Wq  = (const float*)d_in[1];
  const float* Wk  = (const float*)d_in[2];
  const float* Wv  = (const float*)d_in[3];
  const float* Wo  = (const float*)d_in[4];
  const float* rel = (const float*)d_in[5];
  float* out = (float*)d_out;

  const size_t NX = (size_t)B_ * T_ * D_;   // 8388608
  const size_t NW = (size_t)D_ * D_;        // 1048576
  _Float16* Xh = (_Float16*)d_ws;
  _Float16* Wh = Xh + NX;                   // Wq,Wk,Wv,Wo packed f16
  _Float16* Qh = Wh + 4 * NW;               // [B,NH,T,HS], pre-scaled by 1/8
  _Float16* Kh = Qh + NX;                   // [B,NH,T,HS]
  _Float16* Vt = Kh + NX;                   // [B,NH,HS,T]
  _Float16* Oh = Vt + NX;                   // [B,T,D]
  float* btab  = (float*)(Oh + NX);         // [NH][4095]

  const int thr = 256;
  f32_to_f16_kernel<<<(int)((NX + thr - 1) / thr), thr, 0, stream>>>(q,  Xh, (int)NX);
  f32_to_f16_kernel<<<(int)((NW + thr - 1) / thr), thr, 0, stream>>>(Wq, Wh + 0 * NW, (int)NW);
  f32_to_f16_kernel<<<(int)((NW + thr - 1) / thr), thr, 0, stream>>>(Wk, Wh + 1 * NW, (int)NW);
  f32_to_f16_kernel<<<(int)((NW + thr - 1) / thr), thr, 0, stream>>>(Wv, Wh + 2 * NW, (int)NW);
  f32_to_f16_kernel<<<(int)((NW + thr - 1) / thr), thr, 0, stream>>>(Wo, Wh + 3 * NW, (int)NW);
  bias_table_kernel<<<(NH_ * BIAS_W + thr - 1) / thr, thr, 0, stream>>>(rel, btab);

  dim3 g1(D_ / 128, (B_ * T_) / 128, 3);
  gemm_qkvo_kernel<<<g1, 256, 0, stream>>>(Xh, Wh, Qh, Kh, Vt, nullptr, 0);

  dim3 g2(T_ / 64, B_ * NH_, 1);
  attn_kernel<<<g2, 128, 0, stream>>>(Qh, Kh, Vt, btab, Oh);

  dim3 g3(D_ / 128, (B_ * T_) / 128, 1);
  gemm_qkvo_kernel<<<g3, 256, 0, stream>>>(Oh, Wh + 3 * NW, nullptr, nullptr, nullptr, out, 1);
}
